// HierarchicalBrainGNN_4346506903769
// MI455X (gfx1250) — compile-verified, matrix operand order checked by reading, not run on hardware
//
#include <hip/hip_runtime.h>

typedef __attribute__((ext_vector_type(16))) __bf16 v16bf;
typedef __attribute__((ext_vector_type(8)))  float  v8f;

#define N_NODES 20000
#define N_EDGES 320000
#define ETOT    (N_EDGES + N_NODES)   // with self loops
#define NB      8
#define F_IN    100
#define HID     256
#define HEADS   4
#define NEG_SLOPE 0.2f
#define LN_EPS    1e-5f
#define MT      5                      // M tiles (16 rows each) per wave: 80 rows/block

// ---------- helpers ----------

__device__ __forceinline__ unsigned short f2bf(float f) {
  unsigned u = __float_as_uint(f);
  u += 0x7FFFu + ((u >> 16) & 1u);   // round-to-nearest-even
  return (unsigned short)(u >> 16);
}

// monotonic float <-> uint mapping for atomicMax on floats
__device__ __forceinline__ unsigned ordmap(float f) {
  unsigned u = __float_as_uint(f);
  return (u & 0x80000000u) ? ~u : (u | 0x80000000u);
}
__device__ __forceinline__ float orddec(unsigned u) {
  return __uint_as_float((u & 0x80000000u) ? (u ^ 0x80000000u) : ~u);
}

// ---------- fills ----------

__global__ void fill_u32(unsigned* p, unsigned v, long n) {
  long i = (long)blockIdx.x * blockDim.x + threadIdx.x;
  if (i < n) p[i] = v;
}

// ---------- conversions ----------

// x [N,100] f32 -> xpad [N,128] bf16 (zero pad K)
__global__ void pad_x_kernel(const float* x, unsigned short* xp) {
  long i = (long)blockIdx.x * blockDim.x + threadIdx.x;
  if (i >= (long)N_NODES * 128) return;
  int n = (int)(i >> 7), k = (int)(i & 127);
  xp[i] = (k < F_IN) ? f2bf(x[(long)n * F_IN + k]) : (unsigned short)0;
}

// W [Kin,cols] f32 -> [Kpad,cols] bf16, zero pad rows >= Kin
__global__ void conv_w_kernel(const float* w, unsigned short* o, int Kin, int Kpad, int cols) {
  long i = (long)blockIdx.x * blockDim.x + threadIdx.x;
  if (i >= (long)Kpad * cols) return;
  int r = (int)(i / cols), c = (int)(i % cols);
  o[i] = (r < Kin) ? f2bf(w[(long)r * cols + c]) : (unsigned short)0;
}

__global__ void f32_to_bf16_kernel(const float* in, unsigned short* out, long n) {
  long i = (long)blockIdx.x * blockDim.x + threadIdx.x;
  if (i < n) out[i] = f2bf(in[i]);
}

// ---------- WMMA GEMM: C[M,Nc] = A[M,K]bf16 * B[K,Nc]bf16 (+bias, relu) ----------
// Block: 256 threads = 8 waves. Each wave owns one 16-col tile (block covers
// 128 cols exactly; all Nc are multiples of 128). Each wave computes MT=5
// 16-row tiles (80 rows/block; 20000/80 = 250 exact, no guards, EXEC all-1s).
//
// Per K-step the 32x128 B slab is staged in LDS *transposed to fragment-major
// order*: Bs3[col][k] keeps each column's 32 K-values contiguous. The ISA B
// fragment for v_wmma_f32_16x16x32_bf16 is then two contiguous 16B runs per
// lane (K = half*8 + 0..7 and K = 16 + half*8 + 0..7 at a fixed column), so
// the fragment load is 2x ds_load_b128 instead of 16 scalar ds_load_u16.
// The transpose happens on the store side: each thread loads two consecutive
// K rows (coalesced b128) and scatters 8 packed (k,k+1) pairs as ds_store_b32.

union Frag { v16bf v; unsigned short s[16]; uint4 q[2]; };

__global__ void gemm_bf16_wmma(const unsigned short* __restrict__ A, int lda,
                               const unsigned short* __restrict__ B, int ldb,
                               float* __restrict__ C, int ldc,
                               int K, const float* __restrict__ bias, int relu) {
  __shared__ unsigned short Bs3[128][32];           // col-major slab, 8 KB

  const int tid  = threadIdx.x;
  const int lane = tid & 31;
  const int wave = tid >> 5;
  const int half = lane >> 4;                       // 0/1
  const int idx  = lane & 15;
  const int colbase = blockIdx.y * 128;
  const int col  = colbase + wave * 16 + idx;       // B col / C col for this lane
  const int row0 = blockIdx.x * (MT * 16);

  // staging assignment: pair-row u = tid>>4 (0..15), column chunk (tid&15)*8
  const int su = tid >> 4;
  const int sc = (tid & 15) << 3;

  v8f acc[MT] = {};

  for (int k0 = 0; k0 < K; k0 += 32) {
    __syncthreads();                                // prior-iteration reads done
    // load rows k0+2u, k0+2u+1 (8 cols each, coalesced b128); scatter pairs
    union { uint4 q; unsigned short s[8]; } r0, r1;
    r0.q = *(const uint4*)&B[(long)(k0 + 2 * su) * ldb + colbase + sc];
    r1.q = *(const uint4*)&B[(long)(k0 + 2 * su + 1) * ldb + colbase + sc];
#pragma unroll
    for (int j = 0; j < 8; ++j) {
      unsigned pair = (unsigned)r0.s[j] | ((unsigned)r1.s[j] << 16);
      *(unsigned*)&Bs3[sc + j][2 * su] = pair;      // Bs3[col][k]=B[k][col]
    }
    if (k0 + 32 < K)                                // hint next slab into cache
      __builtin_prefetch(&B[(long)(k0 + 32 + lane) * ldb + colbase + (wave << 4)], 0, 1);
    __syncthreads();

    // per-lane B fragment: two contiguous 16B runs (ds_load_b128 x2)
    Frag b;
    b.q[0] = *(const uint4*)&Bs3[wave * 16 + idx][half * 8];
    b.q[1] = *(const uint4*)&Bs3[wave * 16 + idx][16 + half * 8];

    // 5 M-tiles against the shared B fragment
#pragma unroll
    for (int t = 0; t < MT; ++t) {
      int row = row0 + t * 16 + idx;                // A row held by this lane
      Frag a;
      a.q[0] = *(const uint4*)&A[(long)row * lda + k0 + half * 8];
      a.q[1] = *(const uint4*)&A[(long)row * lda + k0 + 16 + half * 8];
      acc[t] = __builtin_amdgcn_wmma_f32_16x16x32_bf16(
          false, a.v, false, b.v, (short)0, acc[t], false, false);
    }
  }

  float bv = bias ? bias[col] : 0.0f;
#pragma unroll
  for (int t = 0; t < MT; ++t) {
#pragma unroll
    for (int j = 0; j < 8; ++j) {
      int r = row0 + t * 16 + j + half * 8;         // C layout: VGPR j -> M=j+8*half
      float v = acc[t][j] + bv;
      if (relu) v = v > 0.0f ? v : 0.0f;
      C[(long)r * ldc + col] = v;
    }
  }
}

// ---------- attention logits per node: al = sum_c feat[n,h,c]*a[h,c] ----------

__global__ void compute_al_kernel(const float* __restrict__ feat, int dout,
                                  const float* __restrict__ as_, const float* __restrict__ ad_,
                                  float* __restrict__ als, float* __restrict__ ald) {
  int i = blockIdx.x * blockDim.x + threadIdx.x;   // n*HEADS + h
  if (i >= N_NODES * HEADS) return;
  int n = i / HEADS, h = i % HEADS;
  const float* f = feat + (long)n * HEADS * dout + (long)h * dout;
  float ss = 0.0f, sd = 0.0f;
  for (int c = 0; c < dout; ++c) {
    float v = f[c];
    ss += v * as_[h * dout + c];
    sd += v * ad_[h * dout + c];
  }
  als[i] = ss; ald[i] = sd;
}

// ---------- edge passes ----------

__device__ __forceinline__ void edge_sd(const int* ei, int i, int& src, int& dst) {
  if (i < N_EDGES) { src = ei[i]; dst = ei[N_EDGES + i]; }
  else             { src = i - N_EDGES; dst = i - N_EDGES; }   // self loop
}

__global__ void edge_max_kernel(const int* __restrict__ ei,
                                const float* __restrict__ als, const float* __restrict__ ald,
                                float* __restrict__ e, unsigned* __restrict__ mbits) {
  int i = blockIdx.x * blockDim.x + threadIdx.x;
  if (i >= ETOT) return;
  int src, dst; edge_sd(ei, i, src, dst);
#pragma unroll
  for (int h = 0; h < HEADS; ++h) {
    float v = als[src * HEADS + h] + ald[dst * HEADS + h];
    v = v > 0.0f ? v : NEG_SLOPE * v;              // leaky relu
    e[(long)i * HEADS + h] = v;
    atomicMax(&mbits[dst * HEADS + h], ordmap(v));
  }
}

__global__ void finalize_m_kernel(unsigned* mb) {
  int i = blockIdx.x * blockDim.x + threadIdx.x;
  if (i >= N_NODES * HEADS) return;
  float m = orddec(mb[i]);
  if (!(m > -3.0e38f && m < 3.0e38f)) m = 0.0f;    // where(isfinite(m), m, 0)
  ((float*)mb)[i] = m;
}

__global__ void edge_expsum_kernel(const int* __restrict__ ei,
                                   float* __restrict__ e, const float* __restrict__ m,
                                   float* __restrict__ den) {
  int i = blockIdx.x * blockDim.x + threadIdx.x;
  if (i >= ETOT) return;
  int src, dst; edge_sd(ei, i, src, dst);
#pragma unroll
  for (int h = 0; h < HEADS; ++h) {
    float ex = __expf(e[(long)i * HEADS + h] - m[dst * HEADS + h]);
    e[(long)i * HEADS + h] = ex;
    atomicAdd(&den[dst * HEADS + h], ex);
  }
}

__global__ void edge_aggregate_kernel(const int* __restrict__ ei,
                                      const float* __restrict__ ex, const float* __restrict__ den,
                                      const float* __restrict__ feat, int dout,
                                      float* __restrict__ outacc) {
  int t = blockIdx.x * blockDim.x + threadIdx.x;   // edge*HEADS + h
  if (t >= ETOT * HEADS) return;
  int i = t / HEADS, h = t % HEADS;
  int src, dst; edge_sd(ei, i, src, dst);
  float alpha = ex[(long)i * HEADS + h] / (den[dst * HEADS + h] + 1e-16f);
  const float* fs = feat + (long)src * HEADS * dout + (long)h * dout;
  float* od = outacc + (long)dst * HEADS * dout + (long)h * dout;
  for (int c = 0; c < dout; ++c)
    atomicAdd(&od[c], alpha * fs[c]);
}

// ---------- node epilogue: head-mean + bias, relu, layernorm, residual ----------

__global__ void node_epilogue_kernel(const float* __restrict__ outacc, int dout,
                                     const float* __restrict__ b,
                                     const float* __restrict__ g,
                                     const float* __restrict__ be,
                                     int resid, int din,
                                     float* __restrict__ hF, unsigned short* __restrict__ hB) {
  int n = blockIdx.x * blockDim.x + threadIdx.x;
  if (n >= N_NODES) return;
  const float* o = outacc + (long)n * HEADS * dout;
  float s = 0.0f, s2 = 0.0f;
  for (int c = 0; c < dout; ++c) {
    float v = 0.25f * (o[c] + o[dout + c] + o[2 * dout + c] + o[3 * dout + c]) + b[c];
    v = v > 0.0f ? v : 0.0f;
    s += v; s2 += v * v;
  }
  float mean = s / dout;
  float var  = s2 / dout - mean * mean;
  float rstd = rsqrtf((var > 0.0f ? var : 0.0f) + LN_EPS);
  for (int c = 0; c < dout; ++c) {
    float v = 0.25f * (o[c] + o[dout + c] + o[2 * dout + c] + o[3 * dout + c]) + b[c];
    v = v > 0.0f ? v : 0.0f;
    float y = (v - mean) * rstd * g[c] + be[c];
    if (resid) y += hF[(long)n * din + c];          // din==dout when resid
    hF[(long)n * dout + c] = y;
    hB[(long)n * dout + c] = f2bf(y);
  }
}

// ---------- pooling + output ----------

__global__ void pool_kernel(const int* __restrict__ batch, const float* __restrict__ h,
                            float* __restrict__ pooled, float* __restrict__ cnt) {
  int n = blockIdx.x * blockDim.x + threadIdx.x;
  if (n >= N_NODES) return;
  int bq = batch[n];
  atomicAdd(&cnt[bq], 1.0f);
  for (int c = 0; c < 32; ++c)
    atomicAdd(&pooled[bq * 32 + c], h[(long)n * 32 + c]);
}

__global__ void final_kernel(const float* __restrict__ pooled, const float* __restrict__ cnt,
                             const float* __restrict__ Wo, const float* __restrict__ bo,
                             float* __restrict__ out) {
  int bq = threadIdx.x;
  if (bq >= NB) return;
  float c = cnt[bq]; if (c < 1.0f) c = 1.0f;
  float inv = 1.0f / c;
  float acc = bo[0];
  for (int k = 0; k < 32; ++k)
    acc += pooled[bq * 32 + k] * inv * Wo[k];
  out[bq] = acc;
}

// ---------- host orchestration ----------

extern "C" void kernel_launch(void* const* d_in, const int* in_sizes, int n_in,
                              void* d_out, int out_size, void* d_ws, size_t ws_size,
                              hipStream_t stream) {
  (void)in_sizes; (void)n_in; (void)out_size; (void)ws_size;

  const float* x     = (const float*)d_in[0];
  const int*   ei    = (const int*)d_in[1];
  const int*   batch = (const int*)d_in[2];
  const float* W_in  = (const float*)d_in[3];
  const float* b_in  = (const float*)d_in[4];
  const float* W_out = (const float*)d_in[29];
  const float* b_out = (const float*)d_in[30];
  float* out = (float*)d_out;

  static const int DI[4] = {256, 256, 128, 64};
  static const int DO[4] = {256, 128, 64, 32};

  char* ws = (char*)d_ws;
  size_t off = 0;
  auto alloc = [&](size_t bytes) -> void* {
    off = (off + 255) & ~(size_t)255;
    void* p = ws + off; off += bytes; return p;
  };
  unsigned short* xpad  = (unsigned short*)alloc((size_t)N_NODES * 128 * 2);
  unsigned short* wbuf  = (unsigned short*)alloc((size_t)256 * 1024 * 2);
  float*          hF    = (float*)         alloc((size_t)N_NODES * 256 * 4);
  unsigned short* hB    = (unsigned short*)alloc((size_t)N_NODES * 256 * 2);
  float*          feat  = (float*)         alloc((size_t)N_NODES * 1024 * 4);
  float*          outac = (float*)         alloc((size_t)N_NODES * 1024 * 4);
  float*          als   = (float*)         alloc((size_t)N_NODES * HEADS * 4);
  float*          ald   = (float*)         alloc((size_t)N_NODES * HEADS * 4);
  float*          ebuf  = (float*)         alloc((size_t)ETOT * HEADS * 4);
  unsigned*       mbits = (unsigned*)      alloc((size_t)N_NODES * HEADS * 4);
  float*          den   = (float*)         alloc((size_t)N_NODES * HEADS * 4);
  float*          pooled= (float*)         alloc((size_t)NB * 32 * 4);
  float*          cnt   = (float*)         alloc((size_t)NB * 4);

  const int T = 256;
  auto blocks = [](long n, int t) { return (unsigned)((n + t - 1) / t); };

  // 1) pad+convert x, convert W_in (100->128 K pad)
  pad_x_kernel<<<blocks((long)N_NODES * 128, T), T, 0, stream>>>(x, xpad);
  conv_w_kernel<<<blocks((long)128 * HID, T), T, 0, stream>>>(W_in, wbuf, F_IN, 128, HID);

  // 2) h = relu(x @ W_in + b_in)   [N,256], K=128
  {
    dim3 grid(N_NODES / (16 * MT), HID / 128);
    gemm_bf16_wmma<<<grid, 256, 0, stream>>>(xpad, 128, wbuf, HID, hF, HID,
                                             128, b_in, 1);
  }
  f32_to_bf16_kernel<<<blocks((long)N_NODES * HID, T), T, 0, stream>>>(hF, hB, (long)N_NODES * HID);

  // 3) GAT levels
  for (int l = 0; l < 4; ++l) {
    const int di = DI[l], dq = DO[l], nc = HEADS * dq;
    const int base = 5 + 6 * l;
    const float* W   = (const float*)d_in[base + 0];
    const float* as_ = (const float*)d_in[base + 1];
    const float* ad_ = (const float*)d_in[base + 2];
    const float* bb  = (const float*)d_in[base + 3];
    const float* gg  = (const float*)d_in[base + 4];
    const float* be  = (const float*)d_in[base + 5];

    // feat = h @ W  [N, H*do]
    conv_w_kernel<<<blocks((long)di * nc, T), T, 0, stream>>>(W, wbuf, di, di, nc);
    {
      dim3 grid(N_NODES / (16 * MT), nc / 128);
      gemm_bf16_wmma<<<grid, 256, 0, stream>>>(hB, di, wbuf, nc, feat, nc,
                                               di, nullptr, 0);
    }

    // attention logits
    compute_al_kernel<<<blocks((long)N_NODES * HEADS, T), T, 0, stream>>>(feat, dq, as_, ad_, als, ald);

    // init segment buffers: m = -inf (ordered), den = 0, outacc = 0
    fill_u32<<<blocks((long)N_NODES * HEADS, T), T, 0, stream>>>(mbits, 0x007FFFFFu, (long)N_NODES * HEADS);
    fill_u32<<<blocks((long)N_NODES * HEADS, T), T, 0, stream>>>((unsigned*)den, 0u, (long)N_NODES * HEADS);
    fill_u32<<<blocks((long)N_NODES * nc, T), T, 0, stream>>>((unsigned*)outac, 0u, (long)N_NODES * nc);

    // segment softmax over incoming edges + aggregate
    edge_max_kernel<<<blocks(ETOT, T), T, 0, stream>>>(ei, als, ald, ebuf, mbits);
    finalize_m_kernel<<<blocks((long)N_NODES * HEADS, T), T, 0, stream>>>(mbits);
    edge_expsum_kernel<<<blocks(ETOT, T), T, 0, stream>>>(ei, ebuf, (const float*)mbits, den);
    edge_aggregate_kernel<<<blocks((long)ETOT * HEADS, T), T, 0, stream>>>(ei, ebuf, den, feat, dq, outac);

    // head mean + bias, relu, layernorm, residual (level 0 only)
    node_epilogue_kernel<<<blocks(N_NODES, T), T, 0, stream>>>(outac, dq, bb, gg, be,
                                                               (di == dq) ? 1 : 0, di, hF, hB);
  }

  // 4) mean pool per graph + linear head
  fill_u32<<<1, 256, 0, stream>>>((unsigned*)pooled, 0u, NB * 32);
  fill_u32<<<1, 32, 0, stream>>>((unsigned*)cnt, 0u, NB);
  pool_kernel<<<blocks(N_NODES, T), T, 0, stream>>>(batch, hF, pooled, cnt);
  final_kernel<<<1, 32, 0, stream>>>(pooled, cnt, W_out, b_out, out);
}
